// NetVLAD_15659450761895
// MI455X (gfx1250) — matrix operand order repaired and checked by reading, not compiled, not run on hardware
//
#include <hip/hip_runtime.h>
#include <math.h>

#define NPOS 4096      // H*W
#define CCH  128       // channels
#define KCL  64        // clusters
#define KE   56        // kept clusters
#define NB   32        // batch
#define EPSN 1e-12f

typedef __attribute__((ext_vector_type(16))) _Float16 v16h;
typedef __attribute__((ext_vector_type(8)))  _Float16 v8h;
typedef __attribute__((ext_vector_type(8)))  float    v8f;

// Build a 16-half WMMA fragment from two contiguous 16-byte chunks.
static __device__ __forceinline__ v16h load_frag2(const _Float16* p0, const _Float16* p1) {
    v8h lo = *(const v8h*)p0;
    v8h hi = *(const v8h*)p1;
    v16h r;
#pragma unroll
    for (int i = 0; i < 8; i++) { r[i] = lo[i]; r[i + 8] = hi[i]; }
    return r;
}

// ---------------------------------------------------------------------------
// conv_w (64x128 f32) -> f16
__global__ void k_prep(const float* __restrict__ cw, _Float16* __restrict__ cw16) {
    int i = blockIdx.x * 256 + threadIdx.x;
    if (i < KCL * CCH) cw16[i] = (_Float16)cw[i];
}

// zero the f32 accumulators (vlad 32*64*128, asum 32*64)
__global__ void k_zero(float* __restrict__ vlad, float* __restrict__ asum) {
    int i = blockIdx.x * 256 + threadIdx.x;
    if (i < NB * KCL * CCH) vlad[i] = 0.0f;
    if (i < NB * KCL)       asum[i] = 0.0f;
}

// ---------------------------------------------------------------------------
// Per-position L2 normalize over C; emit f16 in BOTH layouts:
//   xn  [n][p][c]  (c contiguous; B operand of GEMM1)
//   xnT [n][c][p]  (p contiguous; B operand of GEMM2)
// Block: 256 threads, tile = 64 positions of one image.
__global__ void k_normalize(const float* __restrict__ x,
                            _Float16* __restrict__ xn,
                            _Float16* __restrict__ xnT) {
    __shared__ float tile[CCH * 65];   // [c][p], padded stride 65 vs 64 banks
    __shared__ float psum[4 * 64];
    __shared__ float rn[64];
    const int n  = blockIdx.y;
    const int p0 = blockIdx.x * 64;
    const int tid = threadIdx.x;
    const int tx = tid & 63, ty = tid >> 6;
    const float* xb = x + (size_t)n * CCH * NPOS;
    float ss = 0.0f;
#pragma unroll 4
    for (int c = ty; c < CCH; c += 4) {
        float v = xb[(size_t)c * NPOS + p0 + tx];
        tile[c * 65 + tx] = v;
        ss += v * v;
    }
    psum[ty * 64 + tx] = ss;
    __syncthreads();
    if (ty == 0) {
        float t = psum[tx] + psum[64 + tx] + psum[128 + tx] + psum[192 + tx];
        rn[tx] = 1.0f / fmaxf(sqrtf(t), EPSN);
    }
    __syncthreads();
    // layout 1: [p][c]
    {
        const int c2 = tid & 127, pg = tid >> 7;
        _Float16* ob = xn + ((size_t)n * NPOS + p0) * CCH;
#pragma unroll 8
        for (int i = 0; i < 32; i++) {
            int p = 2 * i + pg;
            ob[(size_t)p * CCH + c2] = (_Float16)(tile[c2 * 65 + p] * rn[p]);
        }
    }
    // layout 2: [c][p]
    {
        _Float16* obT = xnT + (size_t)n * CCH * NPOS + p0;
        const float r = rn[tx];
#pragma unroll 4
        for (int c = ty; c < CCH; c += 4) {
            obT[(size_t)c * NPOS + tx] = (_Float16)(tile[c * 65 + tx] * r);
        }
    }
}

// ---------------------------------------------------------------------------
// GEMM1 + softmax: logits[k,p] = conv_w[k,:]·xn[:,p] + b[k]; softmax over 64 k.
// Each wave: full 64-k column x 16 positions. 4 K-chunks of 32 over C=128.
// Block = 8 waves = 128 positions.  a stored [n][k][p] f16, rows >= KE zeroed.
__global__ void k_logits(const _Float16* __restrict__ xn,
                         const _Float16* __restrict__ cw16,
                         const float* __restrict__ cb,
                         _Float16* __restrict__ aout) {
    __shared__ float sbias[KCL];
    const int n   = blockIdx.y;
    const int tid = threadIdx.x;
    if (tid < KCL) sbias[tid] = cb[tid];
    __syncthreads();
    const int wave = tid >> 5, lane = tid & 31;
    const int lhalf = lane & 15, hi = lane >> 4;
    const int p = blockIdx.x * 128 + wave * 16 + lhalf;

    v8f acc[4] = {};
    const _Float16* xb = xn + ((size_t)n * NPOS + p) * CCH;
#pragma unroll
    for (int kc = 0; kc < 4; kc++) {
        const int cbase = kc * 32;
        // B fragment: elements j = xn[p][cbase + hi*16 + j], contiguous
        v16h bfrag = load_frag2(xb + cbase + hi * 16, xb + cbase + hi * 16 + 8);
#pragma unroll
        for (int m = 0; m < 4; m++) {
            const _Float16* ab = cw16 + (size_t)(m * 16 + lhalf) * CCH + cbase + hi * 16;
            v16h afrag = load_frag2(ab, ab + 8);
            acc[m] = __builtin_amdgcn_wmma_f32_16x16x32_f16(
                false, afrag, false, bfrag, (short)0, acc[m], false, false);
        }
    }
    // softmax over the 64 cluster logits of position p (split across lane, lane^16)
    float ev[4][8];
    float mx = -1e30f;
#pragma unroll
    for (int m = 0; m < 4; m++)
#pragma unroll
        for (int r = 0; r < 8; r++) {
            int k = m * 16 + hi * 8 + r;
            float lg = acc[m][r] + sbias[k];
            ev[m][r] = lg;
            mx = fmaxf(mx, lg);
        }
    mx = fmaxf(mx, __shfl_xor(mx, 16, 32));
    float s = 0.0f;
#pragma unroll
    for (int m = 0; m < 4; m++)
#pragma unroll
        for (int r = 0; r < 8; r++) {
            float e = __expf(ev[m][r] - mx);
            ev[m][r] = e;
            s += e;
        }
    s += __shfl_xor(s, 16, 32);
    const float rinv = 1.0f / s;
#pragma unroll
    for (int m = 0; m < 4; m++)
#pragma unroll
        for (int r = 0; r < 8; r++) {
            int k = m * 16 + hi * 8 + r;
            float v = (k < KE) ? ev[m][r] * rinv : 0.0f;
            aout[((size_t)n * KCL + k) * NPOS + p] = (_Float16)v;
        }
}

// ---------------------------------------------------------------------------
// GEMM2: vlad[k,c] += sum_p a[k,p]*xn[p,c].  K(p) split 8 ways across blocks,
// f32 atomic accumulation.  Wave w covers c-tile w (16 c) x all 64 k.
// Wave 0 additionally folds asum[k] = sum_p a[k,p] out of its A fragments.
__global__ void k_vlad(const _Float16* __restrict__ aout,
                       const _Float16* __restrict__ xnT,
                       float* __restrict__ vlad,
                       float* __restrict__ asum) {
    const int n = blockIdx.y;
    const int split = blockIdx.x;        // 0..7 -> p range of 512
    const int tid = threadIdx.x;
    const int wave = tid >> 5, lane = tid & 31;
    const int lhalf = lane & 15, hi = lane >> 4;
    const int c = wave * 16 + lhalf;

    v8f acc[4] = {};
    float apart[4] = {0.0f, 0.0f, 0.0f, 0.0f};
    const int pend = split * 512 + 512;
    for (int pb = split * 512; pb < pend; pb += 32) {
        // B fragment: elements j = xnT[c][pb + hi*16 + j], contiguous
        const _Float16* xb = xnT + ((size_t)n * CCH + c) * NPOS + pb + hi * 16;
        v16h bfrag = load_frag2(xb, xb + 8);
#pragma unroll
        for (int m = 0; m < 4; m++) {
            const _Float16* ab =
                aout + ((size_t)n * KCL + m * 16 + lhalf) * NPOS + pb + hi * 16;
            v16h afrag = load_frag2(ab, ab + 8);
            if (wave == 0) {
                float s = 0.0f;
#pragma unroll
                for (int j = 0; j < 16; j++) s += (float)afrag[j];
                apart[m] += s;
            }
            acc[m] = __builtin_amdgcn_wmma_f32_16x16x32_f16(
                false, afrag, false, bfrag, (short)0, acc[m], false, false);
        }
    }
#pragma unroll
    for (int m = 0; m < 4; m++)
#pragma unroll
        for (int r = 0; r < 8; r++) {
            int k = m * 16 + hi * 8 + r;
            atomicAdd(&vlad[((size_t)n * KCL + k) * CCH + c], acc[m][r]);
        }
    if (wave == 0) {
#pragma unroll
        for (int m = 0; m < 4; m++) {
            float v = apart[m] + __shfl_xor(apart[m], 16, 32);
            if (hi == 0) atomicAdd(&asum[n * KCL + m * 16 + lhalf], v);
        }
    }
}

// ---------------------------------------------------------------------------
// finalize: v = vlad - asum*cen ; intra-norm over k ; global L2 ; write out
__global__ void k_final(const float* __restrict__ vlad, const float* __restrict__ asum,
                        const float* __restrict__ cen, float* __restrict__ out) {
    __shared__ float sval[KE * CCH];
    __shared__ float spss[2 * CCH];
    __shared__ float scol[CCH];
    __shared__ float sred[256];
    __shared__ float rg;
    const int n = blockIdx.x;
    const int tid = threadIdx.x;
    const int c = tid & 127, h = tid >> 7;
    float ss = 0.0f;
    for (int kk = 0; kk < 28; kk++) {
        int k = h * 28 + kk;
        float v = vlad[((size_t)n * KCL + k) * CCH + c] - asum[n * KCL + k] * cen[k * CCH + c];
        sval[k * CCH + c] = v;
        ss += v * v;
    }
    spss[h * CCH + c] = ss;
    __syncthreads();
    if (h == 0) scol[c] = 1.0f / fmaxf(sqrtf(spss[c] + spss[CCH + c]), EPSN);
    __syncthreads();
    float g = 0.0f;
    for (int kk = 0; kk < 28; kk++) {
        int k = h * 28 + kk;
        float v = sval[k * CCH + c] * scol[c];
        g += v * v;
    }
    sred[tid] = g;
    __syncthreads();
    for (int off = 128; off > 0; off >>= 1) {
        if (tid < off) sred[tid] += sred[tid + off];
        __syncthreads();
    }
    if (tid == 0) rg = 1.0f / fmaxf(sqrtf(sred[0]), EPSN);
    __syncthreads();
    for (int kk = 0; kk < 28; kk++) {
        int k = h * 28 + kk;
        out[(size_t)n * (KE * CCH) + k * CCH + c] = sval[k * CCH + c] * scol[c] * rg;
    }
}

// ---------------------------------------------------------------------------
extern "C" void kernel_launch(void* const* d_in, const int* in_sizes, int n_in,
                              void* d_out, int out_size, void* d_ws, size_t ws_size,
                              hipStream_t stream) {
    const float* x   = (const float*)d_in[0];   // (32,128,64,64)
    const float* cw  = (const float*)d_in[1];   // (64,128)
    const float* cb  = (const float*)d_in[2];   // (64,)
    const float* cen = (const float*)d_in[3];   // (64,128)
    float* out = (float*)d_out;                 // (32, 56*128)

    // workspace layout (bytes)
    const size_t off_xn   = 0;                  // 32 MiB  f16 [n][p][c]
    const size_t off_xnT  = 33554432;           // 32 MiB  f16 [n][c][p]
    const size_t off_aw   = 67108864;           // 16 MiB  f16 [n][k][p]
    const size_t off_vlad = 83886080;           //  1 MiB  f32 [n][k][c]
    const size_t off_asum = 84934656;           //  8 KiB  f32 [n][k]
    const size_t off_cw16 = 84942848;           // 16 KiB  f16 [k][c]
    const size_t need     = 84959232;
    if (ws_size < need) return;                 // workspace too small: bail safely

    char* ws = (char*)d_ws;
    _Float16* xn   = (_Float16*)(ws + off_xn);
    _Float16* xnT  = (_Float16*)(ws + off_xnT);
    _Float16* aw   = (_Float16*)(ws + off_aw);
    float*    vlad = (float*)   (ws + off_vlad);
    float*    asum = (float*)   (ws + off_asum);
    _Float16* cw16 = (_Float16*)(ws + off_cw16);

    k_prep<<<32, 256, 0, stream>>>(cw, cw16);
    k_zero<<<(NB * KCL * CCH + 255) / 256, 256, 0, stream>>>(vlad, asum);
    k_normalize<<<dim3(NPOS / 64, NB), 256, 0, stream>>>(x, xn, xnT);
    k_logits<<<dim3(NPOS / 128, NB), 256, 0, stream>>>(xn, cw16, cb, aw);
    k_vlad<<<dim3(8, NB), 256, 0, stream>>>(aw, xnT, vlad, asum);
    k_final<<<NB, 256, 0, stream>>>(vlad, asum, cen, out);
}
